// LSTMModel_74835510165878
// MI455X (gfx1250) — matrix operand order repaired and checked by reading, not compiled, not run on hardware
//
#include <hip/hip_runtime.h>
#include <hip/hip_bf16.h>
#include <math.h>

typedef __bf16 bf16;
typedef __attribute__((ext_vector_type(16))) __bf16 bf16x16;
typedef __attribute__((ext_vector_type(8)))  __bf16 bf16x8;
typedef __attribute__((ext_vector_type(8)))  float  f32x8;

#define BATCH 256
#define SEQ   512
#define HID   256
#define DIN0  64
#define DINR  512          // 2H
#define G4H   1024         // 4H
#define BS    (BATCH * SEQ)

// ---------- WMMA fragment loaders (bf16, 16x16x32 shapes) ----------
// A-matrix 16x32: lane m = l&15; K = hi*16 .. hi*16+15 contiguous
__device__ __forceinline__ bf16x16 load_a_frag(const bf16* __restrict__ rowbase, int hi) {
  bf16x8 lo = *reinterpret_cast<const bf16x8*>(rowbase + hi * 16);
  bf16x8 hv = *reinterpret_cast<const bf16x8*>(rowbase + hi * 16 + 8);
  bf16x16 v;
#pragma unroll
  for (int i = 0; i < 8; ++i) { v[i] = lo[i]; v[i + 8] = hv[i]; }
  return v;
}
// B-matrix 32x16 (stored as [N,K] row-major): lane n = l&15; K = hi*16 .. hi*16+15 contiguous
__device__ __forceinline__ bf16x16 load_b_frag(const bf16* __restrict__ rowbase, int hi) {
  bf16x8 lo = *reinterpret_cast<const bf16x8*>(rowbase + hi * 16);
  bf16x8 hv = *reinterpret_cast<const bf16x8*>(rowbase + hi * 16 + 8);
  bf16x16 v;
#pragma unroll
  for (int i = 0; i < 8; ++i) { v[i] = lo[i]; v[i + 8] = hv[i]; }
  return v;
}

// ---------- big-GEMM, register tiled:  C[M,N](bf16) = A[M,K]*B[N,K]^T + bias ----------
// 256 threads = 8 waves arranged 2(m) x 4(n). Wave tile = 64x64 (4x4 subtiles of 16x16),
// block tile = 128(M) x 256(N). Per K-chunk: 4 A-frag + 4 B-frag loads feed 16 WMMAs
// (4x data reuse vs the naive 1 tile/wave form).  Requires M%128==0, N%256==0, K%32==0.
__global__ void __launch_bounds__(256)
gemm_bf16_wmma_tiled(const bf16* __restrict__ A, int lda,
                     const bf16* __restrict__ Bm, int ldb,
                     bf16* __restrict__ C, int ldc,
                     const float* __restrict__ bias,
                     int K)
{
  const int lane = threadIdx.x & 31;
  const int wave = threadIdx.x >> 5;
  const int wm = wave >> 2;              // 0..1
  const int wn = wave & 3;               // 0..3
  const int r  = lane & 15;
  const int hi = lane >> 4;

  const int m0 = blockIdx.x * 128 + wm * 64;
  const int n0 = blockIdx.y * 256 + wn * 64;

  const bf16* Arow[4];
  const bf16* Brow[4];
#pragma unroll
  for (int i = 0; i < 4; ++i) {
    Arow[i] = A  + (size_t)(m0 + i * 16 + r) * lda;
    Brow[i] = Bm + (size_t)(n0 + i * 16 + r) * ldb;
  }

  f32x8 acc[4][4];
#pragma unroll
  for (int i = 0; i < 4; ++i)
#pragma unroll
    for (int j = 0; j < 4; ++j)
      acc[i][j] = (f32x8){0.f, 0.f, 0.f, 0.f, 0.f, 0.f, 0.f, 0.f};

  for (int k = 0; k < K; k += 32) {
    bf16x16 a[4], b[4];
#pragma unroll
    for (int i = 0; i < 4; ++i) a[i] = load_a_frag(Arow[i] + k, hi);
#pragma unroll
    for (int j = 0; j < 4; ++j) b[j] = load_b_frag(Brow[j] + k, hi);
#pragma unroll
    for (int i = 0; i < 4; ++i)
#pragma unroll
      for (int j = 0; j < 4; ++j)
        acc[i][j] = __builtin_amdgcn_wmma_f32_16x16x32_bf16(false, a[i], false, b[j],
                                                            (short)0, acc[i][j], false, false);
  }

#pragma unroll
  for (int j = 0; j < 4; ++j) {
    const float bv = bias ? bias[n0 + j * 16 + r] : 0.f;
#pragma unroll
    for (int i = 0; i < 4; ++i) {
#pragma unroll
      for (int e = 0; e < 8; ++e) {
        C[(size_t)(m0 + i * 16 + hi * 8 + e) * ldc + (n0 + j * 16 + r)] =
            (bf16)(acc[i][j][e] + bv);
      }
    }
  }
}

// ---------- small GEMM (FC head):  one 16x16 tile per wave ----------
__global__ void __launch_bounds__(256)
gemm_bf16_wmma(const bf16* __restrict__ A, int lda,
               const bf16* __restrict__ Bm, int ldb,
               bf16* __restrict__ C, int ldc,
               const float* __restrict__ bias,
               int K, int relu)
{
  const int lane = threadIdx.x & 31;
  const int wave = threadIdx.x >> 5;
  const int r    = lane & 15;
  const int hi   = lane >> 4;
  const int m0 = blockIdx.x * 128 + wave * 16;
  const int n0 = blockIdx.y * 16;

  const bf16* Arow = A  + (size_t)(m0 + r) * lda;
  const bf16* Brow = Bm + (size_t)(n0 + r) * ldb;

  f32x8 acc = {0.f, 0.f, 0.f, 0.f, 0.f, 0.f, 0.f, 0.f};
  for (int k = 0; k < K; k += 32) {
    bf16x16 a = load_a_frag(Arow + k, hi);
    bf16x16 b = load_b_frag(Brow + k, hi);
    acc = __builtin_amdgcn_wmma_f32_16x16x32_bf16(false, a, false, b,
                                                  (short)0, acc, false, false);
  }
  const float bv = bias ? bias[n0 + r] : 0.f;
#pragma unroll
  for (int i = 0; i < 8; ++i) {
    float v = acc[i] + bv;
    if (relu) v = fmaxf(v, 0.f);
    C[(size_t)(m0 + hi * 8 + i) * ldc + (n0 + r)] = (bf16)v;
  }
}

// ---------- persistent BiLSTM layer kernel ----------
// 64 blocks x 256 threads = 512 waves = 2 dirs x 16 m-tiles x 16 n-tiles.
// Each block stages its 256 KB slice of w_hh (4 gates x 128 hidden cols x 256 K, bf16)
// into LDS ONCE (reused 512 timesteps), keeps c in registers, loops over t, and
// grid-syncs each step through an agent-scope atomic counter.
__device__ __forceinline__ float sigmoidf_(float x) { return 1.f / (1.f + __expf(-x)); }
__device__ __forceinline__ float tanhf_(float x)    { return 1.f - 2.f / (__expf(2.f * x) + 1.f); }

#define LSTM_LDS_ELEMS (4 * 128 * 256)          // bf16 elements = 256 KB

__global__ void __launch_bounds__(256)
lstm_layer_persistent(const bf16* __restrict__ xg,     // [2][B*S][4H] (bias folded in)
                      const bf16* __restrict__ whh,    // [2][4H][H]
                      bf16* __restrict__ seq_out,      // [B][S][2H]
                      bf16* __restrict__ hbuf,         // [2 pp][2 dir][B][H]
                      unsigned* __restrict__ counter)
{
  extern __shared__ bf16 smem[];                       // [4*128][256]

  const int lane = threadIdx.x & 31;
  const int wv   = (blockIdx.x << 3) + (threadIdx.x >> 5);   // 0..511
  const int dir  = wv >> 8;
  const int rem  = wv & 255;
  const int m0   = (rem >> 4) * 16;
  const int nt   = rem & 15;
  const int n0   = nt * 16;
  const int r    = lane & 15;
  const int hi   = lane >> 4;

  const bf16* xg_d  = xg  + (size_t)dir * BS * G4H;
  const bf16* whh_d = whh + (size_t)dir * G4H * HID;

  // ---- stage this block's w_hh slice into LDS (once; reused for all 512 steps)
  const int colbase = (nt & 8) ? 128 : 0;   // same for all waves in the block
  for (int idx = threadIdx.x; idx < LSTM_LDS_ELEMS / 8; idx += 256) {
    const int lin = idx * 8;
    const int gj  = lin >> 8;               // g*128 + j
    const int kk  = lin & 255;
    const int g   = gj >> 7;
    const int j   = gj & 127;
    *reinterpret_cast<bf16x8*>(smem + gj * 256 + kk) =
        *reinterpret_cast<const bf16x8*>(whh_d + (size_t)(g * HID + colbase + j) * HID + kk);
  }
  __syncthreads();

  const bf16* BsRow[4];
#pragma unroll
  for (int g = 0; g < 4; ++g)
    BsRow[g] = smem + ((g * 128 + (nt & 7) * 16 + r) * 256);

  float c[8];
#pragma unroll
  for (int i = 0; i < 8; ++i) c[i] = 0.f;

  const unsigned nwg = gridDim.x;

  for (int t = 0; t < SEQ; ++t) {
    const int s  = dir ? (SEQ - 1 - t) : t;               // reverse dir walks backwards
    const bf16* hprev = hbuf + (size_t)((t & 1) * 2 + dir) * BATCH * HID;
    bf16*       hnext = hbuf + (size_t)((((t + 1) & 1)) * 2 + dir) * BATCH * HID;

    // init accumulators with precomputed input projection xg (bias already folded)
    f32x8 acc[4];
#pragma unroll
    for (int g = 0; g < 4; ++g) {
#pragma unroll
      for (int i = 0; i < 8; ++i) {
        const int b = m0 + hi * 8 + i;
        acc[g][i] = (float)xg_d[((size_t)b * SEQ + s) * G4H + g * HID + n0 + r];
      }
    }

    // gates += h_prev @ w_hh^T   (K = 256; A from global, B frags from LDS; 32 WMMAs)
    const bf16* Arow = hprev + (size_t)(m0 + r) * HID;
#pragma unroll
    for (int kc = 0; kc < HID; kc += 32) {
      bf16x16 a = load_a_frag(Arow + kc, hi);
#pragma unroll
      for (int g = 0; g < 4; ++g) {
        bf16x16 b = load_b_frag(BsRow[g] + kc, hi);
        acc[g] = __builtin_amdgcn_wmma_f32_16x16x32_bf16(false, a, false, b,
                                                         (short)0, acc[g], false, false);
      }
    }

    // fused LSTM cell (gate order i, f, g, o) on the f32 accumulators
#pragma unroll
    for (int i = 0; i < 8; ++i) {
      const int b = m0 + hi * 8 + i;
      const float ig = sigmoidf_(acc[0][i]);
      const float fg = sigmoidf_(acc[1][i]);
      const float gg = tanhf_(acc[2][i]);
      const float og = sigmoidf_(acc[3][i]);
      c[i] = fg * c[i] + ig * gg;
      const float h = og * tanhf_(c[i]);
      const bf16 hb = (bf16)h;
      hnext[(size_t)b * HID + n0 + r] = hb;
      seq_out[((size_t)b * SEQ + s) * DINR + dir * HID + n0 + r] = hb;
    }

    if (t + 1 < SEQ) {
      // prefetch next step's xg rows while we sit in the barrier
      const int sn = dir ? (SEQ - 2 - t) : (t + 1);
#pragma unroll
      for (int g = 0; g < 4; ++g)
        __builtin_prefetch(&xg_d[((size_t)(m0 + hi * 8) * SEQ + sn) * G4H + g * HID + n0 + r], 0, 1);

      __threadfence();
      __syncthreads();
      if (threadIdx.x == 0) {
        __hip_atomic_fetch_add(counter, 1u, __ATOMIC_RELEASE, __HIP_MEMORY_SCOPE_AGENT);
        const unsigned target = (unsigned)(t + 1) * nwg;
        while (__hip_atomic_load(counter, __ATOMIC_ACQUIRE, __HIP_MEMORY_SCOPE_AGENT) < target)
          __builtin_amdgcn_s_sleep(1);
      }
      __syncthreads();
    }
  }
}

// ---------- small helpers ----------
__global__ void f32_to_bf16_kernel(const float* __restrict__ in, bf16* __restrict__ out, size_t n) {
  size_t i = (size_t)blockIdx.x * blockDim.x + threadIdx.x;
  const size_t stride = (size_t)gridDim.x * blockDim.x;
  for (; i < n; i += stride) out[i] = (bf16)in[i];
}

__global__ void zero_u32_kernel(unsigned* __restrict__ p, size_t nwords) {
  size_t i = (size_t)blockIdx.x * blockDim.x + threadIdx.x;
  const size_t stride = (size_t)gridDim.x * blockDim.x;
  for (; i < nwords; i += stride) p[i] = 0u;
}

// fc3: [256,128] bf16 @ [128] f32 + b -> [256] f32
__global__ void fc3_kernel(const bf16* __restrict__ h2, const float* __restrict__ w,
                           const float* __restrict__ b, float* __restrict__ out) {
  const int i = threadIdx.x;       // 256 threads
  float s = 0.f;
#pragma unroll 4
  for (int k = 0; k < 128; ++k) s += (float)h2[i * 128 + k] * w[k];
  out[i] = s + b[0];
}

// =====================================================================
extern "C" void kernel_launch(void* const* d_in, const int* in_sizes, int n_in,
                              void* d_out, int out_size, void* d_ws, size_t ws_size,
                              hipStream_t stream)
{
  (void)in_sizes; (void)n_in; (void)out_size; (void)ws_size;
  const float* x    = (const float*)d_in[0];
  const float* wih0 = (const float*)d_in[1];
  const float* whh0 = (const float*)d_in[2];
  const float* b0   = (const float*)d_in[3];
  const float* wihr = (const float*)d_in[4];
  const float* whhr = (const float*)d_in[5];
  const float* br   = (const float*)d_in[6];
  const float* fc1w = (const float*)d_in[7];
  const float* fc1b = (const float*)d_in[8];
  const float* fc2w = (const float*)d_in[9];
  const float* fc2b = (const float*)d_in[10];
  const float* fc3w = (const float*)d_in[11];
  const float* fc3b = (const float*)d_in[12];
  float* out = (float*)d_out;

  char* ws = (char*)d_ws;
  size_t off = 0;
  auto take = [&](size_t bytes) -> char* {
    char* p = ws + off;
    off += (bytes + 255) & ~(size_t)255;
    return p;
  };

  unsigned* counter = (unsigned*)take(256);                          // barrier counter
  bf16* hbuf    = (bf16*)take((size_t)2 * 2 * BATCH * HID * 2);      // ping-pong h state
  bf16* xg      = (bf16*)take((size_t)2 * BS * G4H * 2);             // per-layer input proj
  bf16* seqA    = (bf16*)take((size_t)BS * DINR * 2);
  bf16* seqB    = (bf16*)take((size_t)BS * DINR * 2);
  bf16* x_bf    = (bf16*)take((size_t)BS * DIN0 * 2);
  bf16* wih0_bf = (bf16*)take((size_t)2 * G4H * DIN0 * 2);
  bf16* whh0_bf = (bf16*)take((size_t)2 * G4H * HID * 2);
  bf16* wihr_bf = (bf16*)take((size_t)2 * 2 * G4H * DINR * 2);
  bf16* whhr_bf = (bf16*)take((size_t)2 * 2 * G4H * HID * 2);
  bf16* fc1w_bf = (bf16*)take((size_t)HID * DINR * 2);
  bf16* fc2w_bf = (bf16*)take((size_t)128 * HID * 2);
  bf16* h1      = (bf16*)take((size_t)BATCH * HID * 2);
  bf16* h2      = (bf16*)take((size_t)BATCH * 128 * 2);

  auto conv = [&](const float* src, bf16* dst, size_t n) {
    f32_to_bf16_kernel<<<2048, 256, 0, stream>>>(src, dst, n);
  };
  conv(x,    x_bf,    (size_t)BS * DIN0);
  conv(wih0, wih0_bf, (size_t)2 * G4H * DIN0);
  conv(whh0, whh0_bf, (size_t)2 * G4H * HID);
  conv(wihr, wihr_bf, (size_t)2 * 2 * G4H * DINR);
  conv(whhr, whhr_bf, (size_t)2 * 2 * G4H * HID);
  conv(fc1w, fc1w_bf, (size_t)HID * DINR);
  conv(fc2w, fc2w_bf, (size_t)128 * HID);

  const dim3 blk(256);
  const size_t hzero_words = (256 + (size_t)2 * 2 * BATCH * HID * 2) / 4;
  const size_t lstm_lds = (size_t)LSTM_LDS_ELEMS * 2;   // 256 KB dynamic LDS

  // ---- layer 0: input projections (tiled WMMA GEMM), then persistent recurrence
  for (int d = 0; d < 2; ++d) {
    gemm_bf16_wmma_tiled<<<dim3(BS / 128, G4H / 256), blk, 0, stream>>>(
        x_bf, DIN0,
        wih0_bf + (size_t)d * G4H * DIN0, DIN0,
        xg + (size_t)d * BS * G4H, G4H,
        b0 + (size_t)d * G4H, DIN0);
  }
  zero_u32_kernel<<<256, 256, 0, stream>>>(counter, hzero_words);
  lstm_layer_persistent<<<64, 256, lstm_lds, stream>>>(xg, whh0_bf, seqA, hbuf, counter);

  // ---- layers 1..2
  bf16* in_seq = seqA;
  bf16* out_seq = seqB;
  for (int l = 0; l < 2; ++l) {
    for (int d = 0; d < 2; ++d) {
      gemm_bf16_wmma_tiled<<<dim3(BS / 128, G4H / 256), blk, 0, stream>>>(
          in_seq, DINR,
          wihr_bf + ((size_t)l * 2 + d) * G4H * DINR, DINR,
          xg + (size_t)d * BS * G4H, G4H,
          br + ((size_t)l * 2 + d) * G4H, DINR);
    }
    zero_u32_kernel<<<256, 256, 0, stream>>>(counter, hzero_words);
    lstm_layer_persistent<<<64, 256, lstm_lds, stream>>>(
        xg, whhr_bf + (size_t)l * 2 * G4H * HID, out_seq, hbuf, counter);
    bf16* tmp = in_seq; in_seq = out_seq; out_seq = tmp;
  }
  // after 2 swaps, final sequence is in in_seq (== seqA)

  // ---- MLP head: fc1 (last timestep view via lda), fc2, fc3
  gemm_bf16_wmma<<<dim3(BATCH / 128, HID / 16), blk, 0, stream>>>(
      in_seq + (size_t)(SEQ - 1) * DINR, SEQ * DINR,
      fc1w_bf, DINR, h1, HID, fc1b, DINR, 1);

  gemm_bf16_wmma<<<dim3(BATCH / 128, 128 / 16), blk, 0, stream>>>(
      h1, HID, fc2w_bf, HID, h2, 128, fc2b, HID, 1);

  fc3_kernel<<<1, 256, 0, stream>>>(h2, fc3w, fc3b, out);
}
